// HybridAttentionBlock_70042326663691
// MI455X (gfx1250) — compile-verified, matrix operand order checked by reading, not run on hardware
//
#include <hip/hip_runtime.h>
#include <hip/hip_bf16.h>

// ---------------------------------------------------------------------------
// Problem constants (match reference)
// ---------------------------------------------------------------------------
constexpr int B_    = 2;
constexpr int S_    = 2048;
constexpr int E_    = 1024;
constexpr int H_    = 16;
constexpr int D_    = 64;
constexpr int RANK_ = 512;
constexpr int KTOP_ = 409;
constexpr int ROWS_ = B_ * S_;      // 4096 flattened (b,s) rows

typedef __attribute__((ext_vector_type(16))) __bf16 v16bf;
typedef __attribute__((ext_vector_type(8)))  float  v8f;

// ---------------------------------------------------------------------------
// Helpers
// ---------------------------------------------------------------------------
__device__ __forceinline__ unsigned short f2bf(float f) {
    unsigned x = __float_as_uint(f);
    unsigned r = x + 0x7fffu + ((x >> 16) & 1u);   // round-to-nearest-even
    return (unsigned short)(r >> 16);
}

__device__ __forceinline__ unsigned pack2bf(float a, float b) {
    return (unsigned)f2bf(a) | ((unsigned)f2bf(b) << 16);
}

// order-preserving float -> uint key (larger key == larger float)
__device__ __forceinline__ unsigned fkey(float f) {
    unsigned u = __float_as_uint(f);
    return (u & 0x80000000u) ? ~u : (u | 0x80000000u);
}

// CDNA5 16-bit A/B fragment element mapping (wave32, 16x32 / 32x16):
// per-lane: vgpr pair v (0..7), half = lane>>4  ->  k index (always even)
__device__ __forceinline__ int amap_k(int v, int half) {
    return ((v >> 2) << 4) + ((v & 3) << 1) + (half << 3);
}

// Build a 16x32 (A) or 32x16 (B) bf16 fragment from memory where, for a fixed
// row/col index (lane&15), the k dimension is contiguous. 8 packed b32 loads
// (the compiler merges these into global_load_b128 pairs).
__device__ __forceinline__ v16bf load_frag(const unsigned short* base,
                                           int stride_elems, int rc, int half,
                                           int k0) {
    union { v16bf v; unsigned u[8]; } f;
    const unsigned short* p = base + (size_t)rc * stride_elems + k0;
#pragma unroll
    for (int v_ = 0; v_ < 8; ++v_)
        f.u[v_] = *(const unsigned*)(p + amap_k(v_, half));
    return f.v;
}

// A-fragment of probabilities held as f32 in LDS (row = query, k contiguous)
__device__ __forceinline__ v16bf load_frag_p(const float* srow, int k0, int half) {
    union { v16bf v; unsigned u[8]; } f;
#pragma unroll
    for (int v_ = 0; v_ < 8; ++v_) {
        float2 d = *(const float2*)(srow + k0 + amap_k(v_, half));
        f.u[v_] = pack2bf(d.x, d.y);
    }
    return f.v;
}

__device__ __forceinline__ v8f wmma_bf16(v16bf a, v16bf b, v8f c) {
    return __builtin_amdgcn_wmma_f32_16x16x32_bf16(
        false, a, false, b, (short)0, c, false, false);
}

// ---------------------------------------------------------------------------
// Weight convert + transpose: f32 W[fin][fout] -> bf16 Wt[fout][fin]
// ---------------------------------------------------------------------------
__global__ void convT_kernel(const float* __restrict__ in,
                             unsigned short* __restrict__ out,
                             int fin, int fout) {
    int i = blockIdx.x * blockDim.x + threadIdx.x;
    if (i >= fin * fout) return;
    int fi = i / fout, fo = i % fout;
    out[(size_t)fo * fin + fi] = f2bf(in[i]);
}

// ---------------------------------------------------------------------------
// LayerNorm (E=1024), one block per row, bf16 output
// ---------------------------------------------------------------------------
__global__ void ln_kernel(const float* __restrict__ x,
                          const float* __restrict__ g,
                          const float* __restrict__ b,
                          unsigned short* __restrict__ xn) {
    __shared__ float r1[256], r2[256];
    const int row = blockIdx.x, tid = threadIdx.x;
    const float4 v = ((const float4*)(x + (size_t)row * E_))[tid];
    float s1 = v.x + v.y + v.z + v.w;
    float s2 = v.x * v.x + v.y * v.y + v.z * v.z + v.w * v.w;
    r1[tid] = s1; r2[tid] = s2;
    __syncthreads();
#pragma unroll
    for (int o = 128; o > 0; o >>= 1) {
        if (tid < o) { r1[tid] += r1[tid + o]; r2[tid] += r2[tid + o]; }
        __syncthreads();
    }
    const float mean = r1[0] * (1.0f / E_);
    const float var  = r2[0] * (1.0f / E_) - mean * mean;
    const float rstd = rsqrtf(var + 1e-5f);
    const float4 gv = ((const float4*)g)[tid];
    const float4 bv = ((const float4*)b)[tid];
    unsigned short* o = xn + (size_t)row * E_ + 4 * tid;
    o[0] = f2bf((v.x - mean) * rstd * gv.x + bv.x);
    o[1] = f2bf((v.y - mean) * rstd * gv.y + bv.y);
    o[2] = f2bf((v.z - mean) * rstd * gv.z + bv.z);
    o[3] = f2bf((v.w - mean) * rstd * gv.w + bv.w);
}

// ---------------------------------------------------------------------------
// Register-blocked WMMA GEMM: C[M,N] = A[M,K] @ Bt[N,K]^T + bias
// One wave computes a 32x64 C strip: 2 A-frags x 4 B-frags -> 8 wmma per
// k-chunk of 32 (1.5 b128 loads per wmma instead of 4).
// mode 0: bf16 out row-major   mode 1: bf16 out transposed (out[n*M+m])
// mode 2: f32 out + bias + residual (final projection)
// Requires M % 32 == 0, N % 64 == 0.
// ---------------------------------------------------------------------------
__global__ void gemm_bf16_kernel(const unsigned short* __restrict__ A,
                                 const unsigned short* __restrict__ Bt,
                                 const float* __restrict__ bias,
                                 int M, int Kd, int N,
                                 void* __restrict__ out,
                                 const float* __restrict__ resid,
                                 int mode) {
    const int lane = threadIdx.x & 31, wave = threadIdx.x >> 5;
    const int half = lane >> 4, l16 = lane & 15;
    const int NT64 = N >> 6;
    const int strips = (M >> 5) * NT64;
    const int strip = blockIdx.x * (blockDim.x >> 5) + wave;
    if (strip >= strips) return;                      // wave-uniform
    const int wn = strip % NT64, wm = strip / NT64;

    const unsigned short* Abase = A + (size_t)(wm * 32) * Kd;
    const unsigned short* Bbase = Bt + (size_t)(wn * 64) * Kd;

    v8f acc[2][4] = {{{}, {}, {}, {}}, {{}, {}, {}, {}}};
    for (int k0 = 0; k0 < Kd; k0 += 32) {
        // prefetch next k-chunk (emits global_prefetch_b8)
        if (k0 + 32 < Kd) {
            __builtin_prefetch(Abase + (size_t)l16 * Kd + k0 + 32, 0, 1);
            __builtin_prefetch(Bbase + (size_t)l16 * Kd + k0 + 32, 0, 1);
        }
        v16bf a[2], b[4];
#pragma unroll
        for (int mi = 0; mi < 2; ++mi)
            a[mi] = load_frag(Abase + (size_t)(mi * 16) * Kd, Kd, l16, half, k0);
#pragma unroll
        for (int ni = 0; ni < 4; ++ni)
            b[ni] = load_frag(Bbase + (size_t)(ni * 16) * Kd, Kd, l16, half, k0);
#pragma unroll
        for (int mi = 0; mi < 2; ++mi)
#pragma unroll
            for (int ni = 0; ni < 4; ++ni)
                acc[mi][ni] = wmma_bf16(a[mi], b[ni], acc[mi][ni]);
    }

#pragma unroll
    for (int ni = 0; ni < 4; ++ni) {
        const int n = wn * 64 + ni * 16 + l16;
        const float bz = bias ? bias[n] : 0.0f;
#pragma unroll
        for (int mi = 0; mi < 2; ++mi)
#pragma unroll
            for (int v_ = 0; v_ < 8; ++v_) {
                const int m = wm * 32 + mi * 16 + v_ + half * 8;
                const float r = acc[mi][ni][v_] + bz;
                if (mode == 0) {
                    ((unsigned short*)out)[(size_t)m * N + n] = f2bf(r);
                } else if (mode == 1) {
                    ((unsigned short*)out)[(size_t)n * M + m] = f2bf(r);
                } else {
                    ((float*)out)[(size_t)m * N + n] =
                        r + resid[(size_t)m * N + n];
                }
            }
    }
}

// ---------------------------------------------------------------------------
// Fused sparse attention: one block = one (b, h, 16-query tile).
// Full score rows (16 x 2048 f32) staged in dynamic LDS; exact top-K_TOP
// radix-select; softmax; PV with WMMA.
// ---------------------------------------------------------------------------
__global__ void attn_kernel(const unsigned short* __restrict__ Q,
                            const unsigned short* __restrict__ Kmat,
                            const unsigned short* __restrict__ Vt,   // [E, B*S]
                            unsigned short* __restrict__ Out) {
    extern __shared__ char smem[];
    float*    sc   = (float*)smem;                          // 16*2048 f32
    unsigned* hist = (unsigned*)(smem + 16 * S_ * 4);       // 16*16 u32
    float*    fred = (float*)(hist + 256);                  // 16*16 f32
    float*    Olds = (float*)(fred + 256);                  // 16*64 f32

    const int blk = blockIdx.x;
    const int qt = blk & 127, bh = blk >> 7;
    const int h = bh & 15, b = bh >> 4;
    const int q0 = qt * 16;

    const int tid = threadIdx.x;
    const int wave = tid >> 5, lane = tid & 31;
    const int half = lane >> 4, l16 = lane & 15;

    // ---- Phase 1: scores = Q K^T / 8 into LDS -----------------------------
    const unsigned short* Qbase = Q + ((size_t)(b * S_ + q0)) * E_ + h * D_;
    const v16bf qa0 = load_frag(Qbase, E_, l16, half, 0);
    const v16bf qa1 = load_frag(Qbase, E_, l16, half, 32);

    for (int kt = wave; kt < S_ / 16; kt += 8) {
        const int kb = kt * 16;
        const unsigned short* Kbase =
            Kmat + ((size_t)(b * S_ + kb)) * E_ + h * D_;
        v16bf kb0 = load_frag(Kbase, E_, l16, half, 0);
        v16bf kb1 = load_frag(Kbase, E_, l16, half, 32);
        v8f c = {};
        c = wmma_bf16(qa0, kb0, c);
        c = wmma_bf16(qa1, kb1, c);
#pragma unroll
        for (int v_ = 0; v_ < 8; ++v_)
            sc[(size_t)(v_ + half * 8) * S_ + kb + l16] = c[v_] * 0.125f;
    }
    // zero O accumulator while we are here
    for (int ii = tid; ii < 16 * D_; ii += 256) Olds[ii] = 0.0f;
    __syncthreads();

    // ---- Phase 2: exact K_TOP-th largest via 8-pass radix select ----------
    const int r = tid >> 4, j = tid & 15;       // 16 threads per query row
    unsigned prefix = 0, kr = KTOP_;
#pragma unroll
    for (int p = 7; p >= 0; --p) {
        hist[r * 16 + j] = 0;
        __syncthreads();
        const int sh = p * 4;
        const unsigned upmask = (p == 7) ? 0u : (0xFFFFFFFFu << (sh + 4));
        for (int i = j; i < S_; i += 16) {
            unsigned u = fkey(sc[(size_t)r * S_ + i]);
            if (((u ^ prefix) & upmask) == 0u)
                atomicAdd(&hist[r * 16 + ((u >> sh) & 15u)], 1u);
        }
        __syncthreads();
        unsigned cnt = 0; int bsel = 0;
        for (int bb = 15; bb >= 0; --bb) {
            unsigned c = hist[r * 16 + bb];
            if (cnt + c >= kr) { bsel = bb; break; }
            cnt += c;
        }
        kr -= cnt;
        prefix |= ((unsigned)bsel) << sh;
        __syncthreads();
    }
    const unsigned thr = prefix;                // key of the 409th largest

    // ---- Phase 3: softmax over kept entries, write P in place -------------
    float mx = -3.4e38f;
    for (int i = j; i < S_; i += 16) mx = fmaxf(mx, sc[(size_t)r * S_ + i]);
    fred[r * 16 + j] = mx;
    __syncthreads();
    mx = fred[r * 16];
#pragma unroll
    for (int t2 = 1; t2 < 16; ++t2) mx = fmaxf(mx, fred[r * 16 + t2]);
    __syncthreads();
    float sum = 0.0f;
    for (int i = j; i < S_; i += 16) {
        float s = sc[(size_t)r * S_ + i];
        if (fkey(s) >= thr) sum += __expf(s - mx);
    }
    fred[r * 16 + j] = sum;
    __syncthreads();
    sum = 0.0f;
#pragma unroll
    for (int t2 = 0; t2 < 16; ++t2) sum += fred[r * 16 + t2];
    const float inv = 1.0f / sum;
    for (int i = j; i < S_; i += 16) {
        float s = sc[(size_t)r * S_ + i];
        sc[(size_t)r * S_ + i] = (fkey(s) >= thr) ? __expf(s - mx) * inv : 0.0f;
    }
    __syncthreads();

    // ---- Phase 4: O = P @ V, keys split across the 8 waves ----------------
    v8f acc[4] = {{}, {}, {}, {}};
    const float* prow = sc + (size_t)l16 * S_;
    for (int kc = wave * 256; kc < wave * 256 + 256; kc += 32) {
        v16bf pa = load_frag_p(prow, kc, half);
#pragma unroll
        for (int dt = 0; dt < 4; ++dt) {
            const unsigned short* Vb =
                Vt + (size_t)(h * D_ + dt * 16) * ROWS_ + (size_t)b * S_;
            v16bf vb = load_frag(Vb, ROWS_, l16, half, kc);
            acc[dt] = wmma_bf16(pa, vb, acc[dt]);
        }
    }
#pragma unroll
    for (int dt = 0; dt < 4; ++dt)
#pragma unroll
        for (int v_ = 0; v_ < 8; ++v_)
            atomicAdd(&Olds[(v_ + half * 8) * D_ + dt * 16 + l16],
                      acc[dt][v_]);
    __syncthreads();

    // ---- Phase 5: write bf16 O tile ---------------------------------------
    for (int ii = tid; ii < 16 * D_; ii += 256) {
        const int m = ii >> 6, d = ii & 63;
        Out[((size_t)(b * S_ + q0 + m)) * E_ + h * D_ + d] = f2bf(Olds[ii]);
    }
}

// ---------------------------------------------------------------------------
// Host launch
// ---------------------------------------------------------------------------
extern "C" void kernel_launch(void* const* d_in, const int* in_sizes, int n_in,
                              void* d_out, int out_size, void* d_ws,
                              size_t ws_size, hipStream_t stream) {
    (void)in_sizes; (void)n_in; (void)out_size; (void)ws_size;

    const float* x    = (const float*)d_in[0];
    const float* qw1  = (const float*)d_in[1];
    const float* qb1  = (const float*)d_in[2];
    const float* qw2  = (const float*)d_in[3];
    const float* qb2  = (const float*)d_in[4];
    const float* kw1  = (const float*)d_in[5];
    const float* kb1  = (const float*)d_in[6];
    const float* kw2  = (const float*)d_in[7];
    const float* kb2  = (const float*)d_in[8];
    const float* vw1  = (const float*)d_in[9];
    const float* vb1  = (const float*)d_in[10];
    const float* vw2  = (const float*)d_in[11];
    const float* vb2  = (const float*)d_in[12];
    const float* ow   = (const float*)d_in[13];
    const float* ob   = (const float*)d_in[14];
    const float* ln_g = (const float*)d_in[15];
    const float* ln_b = (const float*)d_in[16];

    char* ws = (char*)d_ws;
    size_t off = 0;
    auto alloc = [&](size_t bytes) -> void* {
        void* p = ws + off;
        off = (off + bytes + 255) & ~(size_t)255;
        return p;
    };

    unsigned short* xn  = (unsigned short*)alloc((size_t)ROWS_ * E_ * 2);
    unsigned short* tA  = (unsigned short*)alloc((size_t)ROWS_ * RANK_ * 2);
    unsigned short* Qb  = (unsigned short*)alloc((size_t)ROWS_ * E_ * 2);
    unsigned short* Kb  = (unsigned short*)alloc((size_t)ROWS_ * E_ * 2);
    unsigned short* Vt  = (unsigned short*)alloc((size_t)ROWS_ * E_ * 2);
    unsigned short* AO  = (unsigned short*)alloc((size_t)ROWS_ * E_ * 2);
    unsigned short* qw1t = (unsigned short*)alloc((size_t)E_ * RANK_ * 2);
    unsigned short* qw2t = (unsigned short*)alloc((size_t)E_ * RANK_ * 2);
    unsigned short* kw1t = (unsigned short*)alloc((size_t)E_ * RANK_ * 2);
    unsigned short* kw2t = (unsigned short*)alloc((size_t)E_ * RANK_ * 2);
    unsigned short* vw1t = (unsigned short*)alloc((size_t)E_ * RANK_ * 2);
    unsigned short* vw2t = (unsigned short*)alloc((size_t)E_ * RANK_ * 2);
    unsigned short* owt  = (unsigned short*)alloc((size_t)E_ * E_ * 2);

    const int cvt_blk = 256;
    const int n_w1 = E_ * RANK_, n_ow = E_ * E_;
    convT_kernel<<<(n_w1 + cvt_blk - 1) / cvt_blk, cvt_blk, 0, stream>>>(qw1, qw1t, E_, RANK_);
    convT_kernel<<<(n_w1 + cvt_blk - 1) / cvt_blk, cvt_blk, 0, stream>>>(kw1, kw1t, E_, RANK_);
    convT_kernel<<<(n_w1 + cvt_blk - 1) / cvt_blk, cvt_blk, 0, stream>>>(vw1, vw1t, E_, RANK_);
    convT_kernel<<<(n_w1 + cvt_blk - 1) / cvt_blk, cvt_blk, 0, stream>>>(qw2, qw2t, RANK_, E_);
    convT_kernel<<<(n_w1 + cvt_blk - 1) / cvt_blk, cvt_blk, 0, stream>>>(kw2, kw2t, RANK_, E_);
    convT_kernel<<<(n_w1 + cvt_blk - 1) / cvt_blk, cvt_blk, 0, stream>>>(vw2, vw2t, RANK_, E_);
    convT_kernel<<<(n_ow + cvt_blk - 1) / cvt_blk, cvt_blk, 0, stream>>>(ow, owt, E_, E_);

    ln_kernel<<<ROWS_, 256, 0, stream>>>(x, ln_g, ln_b, xn);

    auto gemm = [&](const unsigned short* A, const unsigned short* Bt,
                    const float* bias, int M, int Kd, int N, void* out,
                    const float* resid, int mode) {
        const int strips = (M >> 5) * (N >> 6);   // 32x64 C strip per wave
        gemm_bf16_kernel<<<(strips + 7) / 8, 256, 0, stream>>>(
            A, Bt, bias, M, Kd, N, out, resid, mode);
    };

    // Q = (xn @ qw1 + qb1) @ qw2 + qb2   (bf16, row-major)
    gemm(xn, qw1t, qb1, ROWS_, E_, RANK_, tA, nullptr, 0);
    gemm(tA, qw2t, qb2, ROWS_, RANK_, E_, Qb, nullptr, 0);
    // K
    gemm(xn, kw1t, kb1, ROWS_, E_, RANK_, tA, nullptr, 0);
    gemm(tA, kw2t, kb2, ROWS_, RANK_, E_, Kb, nullptr, 0);
    // V (stored transposed [E, B*S] so PV B-fragments are k-contiguous)
    gemm(xn, vw1t, vb1, ROWS_, E_, RANK_, tA, nullptr, 0);
    gemm(tA, vw2t, vb2, ROWS_, RANK_, E_, Vt, nullptr, 1);

    // Fused sparse attention
    const int smem_bytes = 16 * S_ * 4 + 256 * 4 + 256 * 4 + 16 * D_ * 4;
    hipFuncSetAttribute((const void*)attn_kernel,
                        hipFuncAttributeMaxDynamicSharedMemorySize, smem_bytes);
    attn_kernel<<<B_ * H_ * (S_ / 16), 256, smem_bytes, stream>>>(Qb, Kb, Vt, AO);

    // out = AO @ ow + ob + x   (f32, straight into d_out)
    gemm(AO, owt, ob, ROWS_, E_, E_, d_out, x, 2);
}
